// HydraMamba_45732811768023
// MI455X (gfx1250) — matrix operand order repaired and checked by reading, not compiled, not run on hardware
//
#include <hip/hip_runtime.h>
#include <cstdint>

// ---------------------------------------------------------------------------
// Problem constants (from reference)
// ---------------------------------------------------------------------------
#define SEQ     4096
#define DMODEL  1024
#define DINNER  2048
#define NH      32
#define HD      64
#define DSTATE  128
#define CHUNKL  128
#define NCHUNK  32
#define CONVDIM 2560
#define DPROJ   4672

typedef __bf16 bh;
typedef __attribute__((ext_vector_type(16))) __bf16 v16bf;
typedef __attribute__((ext_vector_type(8)))  __bf16 bf16x8;
typedef __attribute__((ext_vector_type(8)))  float   v8f;

// ---------------------------------------------------------------------------
// helpers
// ---------------------------------------------------------------------------
__device__ inline bh f2bf(float f) {
  union { float f; uint32_t u; } v; v.f = f;
  uint32_t r = v.u + 0x7FFFu + ((v.u >> 16) & 1u);
  uint16_t h = (uint16_t)(r >> 16);
  return __builtin_bit_cast(bh, h);
}
__device__ inline float bf2f(bh b) {
  uint16_t h = __builtin_bit_cast(uint16_t, b);
  union { uint32_t u; float f; } v; v.u = ((uint32_t)h) << 16; return v.f;
}

// A-operand fragment (16x32 bf16) from LDS laid out [m][k], row stride `ld`
// (halfwords, multiple of 8 so 16B loads stay aligned). Per ISA 7.12.2:
// lanes 0-15 hold row M=lane, K = 0..7 and 16..23; lanes 16-31 K = 8..15, 24..31.
__device__ inline v16bf frag_a(const bh* s, int ld, int lane) {
  int r  = lane & 15;
  int kb = (lane >> 4) * 8;
  const bh* p = s + r * ld + kb;
  bf16x8 lo = *(const bf16x8*)(p);        // K = kb .. kb+7
  bf16x8 hi = *(const bf16x8*)(p + 16);   // K = kb+16 .. kb+23
  v16bf o;
#pragma unroll
  for (int j = 0; j < 8; ++j) { o[j] = lo[j]; o[j + 8] = hi[j]; }
  return o;
}

// B-operand fragment (32x16 bf16, K x N) from LDS laid out [n][k].
// lanes 0-15: col N=lane, K = 0..7 and 16..23; lanes 16-31: K = 8..15, 24..31.
__device__ inline v16bf frag_b(const bh* s, int ld, int lane) {
  int n  = lane & 15;
  int kh = (lane >> 4) * 8;
  const bh* p = s + n * ld + kh;
  bf16x8 lo = *(const bf16x8*)(p);
  bf16x8 hi = *(const bf16x8*)(p + 16);
  v16bf o;
#pragma unroll
  for (int j = 0; j < 8; ++j) { o[j] = lo[j]; o[j + 8] = hi[j]; }
  return o;
}

__device__ inline v8f wmma_bf16(v16bf a, v16bf b, v8f c) {
  return __builtin_amdgcn_wmma_f32_16x16x32_bf16(false, a, false, b, (short)0, c,
                                                 false, false);
}

// ---------------------------------------------------------------------------
// generic f32 -> bf16 strided cast
// ---------------------------------------------------------------------------
__global__ __launch_bounds__(256) void k_cast(const float* __restrict__ src,
                                              bh* __restrict__ dst,
                                              int rows, int cols, int sld, int dld) {
  long long i = (long long)blockIdx.x * 256 + threadIdx.x;
  if (i >= (long long)rows * cols) return;
  int c = (int)(i % cols);
  long long r = i / cols;
  dst[r * dld + c] = f2bf(src[r * sld + c]);
}

// ---------------------------------------------------------------------------
// batched GEMM  C(M,N) f32 = A(M,K) bf16 @ B(N,K)^T bf16
// block tile 128x128, BK=32, 8 wave32 waves (4x2), wave tile 32x64 (2x4 WMMA).
// Software-pipelined: double LDS buffers, next tile staged in registers while
// WMMAs consume the current one; one barrier per K-step; L2 prefetch 2 ahead.
// requires K % 32 == 0, lda/ldb multiples of 16 halfwords
// ---------------------------------------------------------------------------
__global__ __launch_bounds__(256) void k_gemm_nt(
    const bh* __restrict__ A, const bh* __restrict__ B, float* __restrict__ C,
    int M, int N, int K, int lda, int ldb, int ldc,
    long long sA, long long sB, long long sC) {
  const int bz = blockIdx.z;
  A += (long long)bz * sA;
  B += (long long)bz * sB;
  C += (long long)bz * sC;
  const int bm = blockIdx.y * 128, bn = blockIdx.x * 128;
  __shared__ alignas(16) bh As[2][128][40];
  __shared__ alignas(16) bh Bs[2][128][40];
  const int tid = threadIdx.x, lane = tid & 31, wave = tid >> 5;
  const int wm = wave >> 1, wn = wave & 1;
  const int r = tid >> 1, c0 = (tid & 1) * 16;
  const int grA = bm + r, grB = bn + r;
  const bh* pA = A + (long long)grA * lda + c0;
  const bh* pB = B + (long long)grB * ldb + c0;
  const bool okA = (grA < M), okB = (grB < N);
  bf16x8 z8 = {};
  v8f acc[2][4] = {};
  const int nk = K >> 5;

  // preload tile 0 into buffer 0
  {
    bf16x8 a0 = z8, a1 = z8, b0 = z8, b1 = z8;
    if (okA) { a0 = *(const bf16x8*)(pA); a1 = *(const bf16x8*)(pA + 8); }
    if (okB) { b0 = *(const bf16x8*)(pB); b1 = *(const bf16x8*)(pB + 8); }
    *(bf16x8*)&As[0][r][c0]     = a0;
    *(bf16x8*)&As[0][r][c0 + 8] = a1;
    *(bf16x8*)&Bs[0][r][c0]     = b0;
    *(bf16x8*)&Bs[0][r][c0 + 8] = b1;
  }

  for (int kt = 0; kt < nk; ++kt) {
    const int cb = kt & 1;
    const bool more = (kt + 1 < nk);
    bf16x8 a0 = z8, a1 = z8, b0 = z8, b1 = z8;
    if (more) {
      const bh* qA = pA + (kt + 1) * 32;
      const bh* qB = pB + (kt + 1) * 32;
      if (okA) { a0 = *(const bf16x8*)(qA); a1 = *(const bf16x8*)(qA + 8); }
      if (okB) { b0 = *(const bf16x8*)(qB); b1 = *(const bf16x8*)(qB + 8); }
      if (kt + 2 < nk) {
        if (okA) __builtin_prefetch(pA + (kt + 2) * 32, 0, 1);
        if (okB) __builtin_prefetch(pB + (kt + 2) * 32, 0, 1);
      }
    }
    __syncthreads();   // tile `cb` is ready; previous reads of `cb^1` are done
    v16bf af[2], bfr[4];
#pragma unroll
    for (int i = 0; i < 2; ++i) af[i] = frag_a(&As[cb][wm * 32 + i * 16][0], 40, lane);
#pragma unroll
    for (int j = 0; j < 4; ++j) bfr[j] = frag_b(&Bs[cb][wn * 64 + j * 16][0], 40, lane);
#pragma unroll
    for (int i = 0; i < 2; ++i)
#pragma unroll
      for (int j = 0; j < 4; ++j) acc[i][j] = wmma_bf16(af[i], bfr[j], acc[i][j]);
    if (more) {
      const int nb = cb ^ 1;
      *(bf16x8*)&As[nb][r][c0]     = a0;
      *(bf16x8*)&As[nb][r][c0 + 8] = a1;
      *(bf16x8*)&Bs[nb][r][c0]     = b0;
      *(bf16x8*)&Bs[nb][r][c0 + 8] = b1;
    }
  }

#pragma unroll
  for (int i = 0; i < 2; ++i)
#pragma unroll
    for (int j = 0; j < 4; ++j) {
      int col = bn + wn * 64 + j * 16 + (lane & 15);
      if (col >= N) continue;
#pragma unroll
      for (int r8 = 0; r8 < 8; ++r8) {
        int row = bm + wm * 32 + i * 16 + r8 + 8 * (lane >> 4);
        if (row < M) C[(long long)row * ldc + col] = acc[i][j][r8];
      }
    }
}

// ---------------------------------------------------------------------------
// depthwise causal conv (width 4) + bias + silu on xBC channels of zxbcdt
// ---------------------------------------------------------------------------
__global__ __launch_bounds__(256) void k_conv(const float* __restrict__ zx,
                                              const float* __restrict__ cw,
                                              const float* __restrict__ cb,
                                              float* __restrict__ xact) {
  long long i = (long long)blockIdx.x * 256 + threadIdx.x;
  if (i >= (long long)SEQ * CONVDIM) return;
  int ch = (int)(i % CONVDIM);
  long long t = i / CONVDIM;
  float acc = cb[ch];
#pragma unroll
  for (int w = 0; w < 4; ++w) {
    long long ts = t - 3 + w;
    if (ts >= 0) acc += cw[ch * 4 + w] * zx[ts * DPROJ + DINNER + ch];
  }
  xact[i] = acc / (1.f + expf(-acc));   // silu
}

// ---------------------------------------------------------------------------
// dt = softplus(dt_raw + bias); dtA = -exp(A_log)*dt.  dir1 seq-flipped.
// layout: (dir, t, h)
// ---------------------------------------------------------------------------
__global__ __launch_bounds__(256) void k_dt(const float* __restrict__ zx,
                                            const float* __restrict__ dt_bias,
                                            const float* __restrict__ A_log,
                                            float* __restrict__ dtb,
                                            float* __restrict__ dtA) {
  long long i = (long long)blockIdx.x * 256 + threadIdx.x;
  if (i >= 2LL * SEQ * NH) return;
  int h = (int)(i & 31);
  long long rr = i >> 5;
  int t = (int)(rr & (SEQ - 1));
  int dir = (int)(rr >> 12);
  int trow = dir ? (SEQ - 1 - t) : t;
  float raw = zx[(long long)trow * DPROJ + (DINNER + CONVDIM) + dir * 32 + h];
  float x = raw + dt_bias[h];
  float sp = (x > 20.f) ? x : log1pf(expf(x));
  dtb[i] = sp;
  dtA[i] = -expf(A_log[h]) * sp;
}

// ---------------------------------------------------------------------------
// per (dir,h,chunk) cumsum of dtA along chunk; also chunk totals
// ---------------------------------------------------------------------------
__global__ __launch_bounds__(256) void k_cumsum(const float* __restrict__ dtA,
                                                float* __restrict__ Acs,
                                                float* __restrict__ csum) {
  int i = blockIdx.x * 256 + threadIdx.x;
  if (i >= 2 * NH * NCHUNK) return;
  int c = i & 31, h = (i >> 5) & 31, dir = i >> 10;
  float run = 0.f;
  long long base = ((long long)(dir * 32 + h) * 32 + c) * 128;
  for (int l = 0; l < CHUNKL; ++l) {
    run += dtA[((long long)dir * SEQ + c * 128 + l) * 32 + h];
    Acs[base + l] = run;
  }
  csum[(dir * 32 + h) * 32 + c] = run;
}

// ---------------------------------------------------------------------------
// build per-direction xdt (bf16), B (bf16), C (bf16) arrays
// ---------------------------------------------------------------------------
__global__ __launch_bounds__(256) void k_arrays(const float* __restrict__ xact,
                                                const float* __restrict__ dtb,
                                                bh* __restrict__ xdt,
                                                bh* __restrict__ Bb,
                                                bh* __restrict__ Cb) {
  int t = blockIdx.x, dir = blockIdx.y, tid = threadIdx.x;
  int tsrc = dir ? (SEQ - 1 - t) : t;
  const float* xr = xact + (long long)tsrc * CONVDIM;
  long long ob = (long long)dir * SEQ + t;
#pragma unroll
  for (int k = 0; k < 8; ++k) {
    int d = tid + k * 256;
    int h = d >> 6;
    float dtv = dtb[ob * 32 + h];
    xdt[ob * 2048 + d] = f2bf(xr[d] * dtv);
  }
  if (tid < 128) {
    int n = tid;
    float Bv = dir ? xr[2304 + n] : xr[2048 + n];
    float Cv = dir ? xr[2432 + n] : xr[2176 + n];
    Bb[ob * 128 + n] = f2bf(Bv);
    Cb[ob * 128 + n] = f2bf(Cv);
  }
}

// ---------------------------------------------------------------------------
// chunk states: S[n,p] = sum_l B[l,n] * exp(Acs[last]-Acs[l]) * xdt[l,p]
// one block per (h, c, dir); WMMA 128x64, K=128
// ---------------------------------------------------------------------------
__global__ __launch_bounds__(256) void k_states(const bh* __restrict__ Bb,
                                                const bh* __restrict__ xdt,
                                                const float* __restrict__ Acs,
                                                float* __restrict__ Sb) {
  int h = blockIdx.x, c = blockIdx.y, dir = blockIdx.z;
  int tid = threadIdx.x, lane = tid & 31, wave = tid >> 5;
  int wm = wave >> 1, wn = wave & 1;
  __shared__ alignas(16) bh SA[128][136];
  __shared__ alignas(16) bh SB[64][136];
  __shared__ float sAcs[128];
  if (tid < 128) sAcs[tid] = Acs[((long long)(dir * 32 + h) * 32 + c) * 128 + tid];
  __syncthreads();
  float last = sAcs[127];
  long long tb = (long long)dir * SEQ + c * 128;
#pragma unroll 4
  for (int it = 0; it < 64; ++it) {              // SA[n][l] = B[l][n]*decay[l]
    int idx = it * 256 + tid;
    int l = idx >> 7, n = idx & 127;
    float bv = bf2f(Bb[(tb + l) * 128 + n]);
    SA[n][l] = f2bf(bv * expf(last - sAcs[l]));
  }
#pragma unroll 4
  for (int it = 0; it < 32; ++it) {              // SB[p][l] = xdt[l][p]
    int idx = it * 256 + tid;
    int l = idx >> 6, p = idx & 63;
    SB[p][l] = xdt[(tb + l) * 2048 + h * 64 + p];
  }
  __syncthreads();
  v8f acc[2][2] = {};
#pragma unroll
  for (int k0 = 0; k0 < 128; k0 += 32) {
    v16bf af[2], bfr[2];
#pragma unroll
    for (int i = 0; i < 2; ++i) af[i] = frag_a(&SA[wm * 32 + i * 16][k0], 136, lane);
#pragma unroll
    for (int j = 0; j < 2; ++j) bfr[j] = frag_b(&SB[wn * 32 + j * 16][k0], 136, lane);
#pragma unroll
    for (int i = 0; i < 2; ++i)
#pragma unroll
      for (int j = 0; j < 2; ++j) acc[i][j] = wmma_bf16(af[i], bfr[j], acc[i][j]);
  }
  long long ob = ((long long)((dir * 32 + c) * 32 + h)) * 8192;
#pragma unroll
  for (int i = 0; i < 2; ++i)
#pragma unroll
    for (int j = 0; j < 2; ++j) {
      int col = wn * 32 + j * 16 + (lane & 15);
#pragma unroll
      for (int r8 = 0; r8 < 8; ++r8) {
        int row = wm * 32 + i * 16 + r8 + 8 * (lane >> 4);
        Sb[ob + (long long)row * 64 + col] = acc[i][j][r8];
      }
    }
}

// ---------------------------------------------------------------------------
// inter-chunk scan: R_{c+1} = R_c * exp(csum[c]) + S_c ; Sin[c] = R_c
// one block per (h, dir); 32 floats per thread in registers
// ---------------------------------------------------------------------------
__global__ __launch_bounds__(256) void k_scan(const float* __restrict__ Sb,
                                              const float* __restrict__ csum,
                                              float* __restrict__ Sin) {
  int h = blockIdx.x, dir = blockIdx.y, tid = threadIdx.x;
  float r[32];
#pragma unroll
  for (int j = 0; j < 32; ++j) r[j] = 0.f;
  int base = tid * 32;
  for (int c = 0; c < NCHUNK; ++c) {
    long long o = ((long long)((dir * 32 + c) * 32 + h)) * 8192 + base;
#pragma unroll
    for (int j = 0; j < 32; ++j) Sin[o + j] = r[j];
    float e = expf(csum[(dir * 32 + h) * 32 + c]);
#pragma unroll
    for (int j = 0; j < 32; ++j) r[j] = r[j] * e + Sb[o + j];
  }
}

// ---------------------------------------------------------------------------
// Y = (G o L) @ xdt + (C o exp(Acs)) @ Sin + x*D
// one block per (h, c, dir); two WMMA accumulation phases into same acc
// ---------------------------------------------------------------------------
__global__ __launch_bounds__(256) void k_Y(const float* __restrict__ G,
                                           const bh* __restrict__ Cb,
                                           const bh* __restrict__ xdt,
                                           const float* __restrict__ Acs,
                                           const float* __restrict__ Sin,
                                           const float* __restrict__ xact,
                                           const float* __restrict__ Dv,
                                           float* __restrict__ Yb) {
  int h = blockIdx.x, c = blockIdx.y, dir = blockIdx.z;
  int tid = threadIdx.x, lane = tid & 31, wave = tid >> 5;
  int wm = wave >> 1, wn = wave & 1;
  __shared__ alignas(16) bh SA[128][136];
  __shared__ alignas(16) bh SB[64][136];
  __shared__ float sAcs[128];
  if (tid < 128) sAcs[tid] = Acs[((long long)(dir * 32 + h) * 32 + c) * 128 + tid];
  __syncthreads();
  long long tb = (long long)dir * SEQ + c * 128;
  const float* Gp = G + ((long long)dir * 32 + c) * 16384;

  // phase 1: Y_diag operands
#pragma unroll 4
  for (int it = 0; it < 64; ++it) {              // SA[l][s] = G*L
    int idx = it * 256 + tid;
    int l = idx >> 7, s = idx & 127;
    float g = (s <= l) ? Gp[l * 128 + s] * expf(sAcs[l] - sAcs[s]) : 0.f;
    SA[l][s] = f2bf(g);
  }
#pragma unroll 4
  for (int it = 0; it < 32; ++it) {              // SB[p][s] = xdt[s][p]
    int idx = it * 256 + tid;
    int s = idx >> 6, p = idx & 63;
    SB[p][s] = xdt[(tb + s) * 2048 + h * 64 + p];
  }
  __syncthreads();
  v8f acc[2][2] = {};
#pragma unroll
  for (int k0 = 0; k0 < 128; k0 += 32) {
    v16bf af[2], bfr[2];
#pragma unroll
    for (int i = 0; i < 2; ++i) af[i] = frag_a(&SA[wm * 32 + i * 16][k0], 136, lane);
#pragma unroll
    for (int j = 0; j < 2; ++j) bfr[j] = frag_b(&SB[wn * 32 + j * 16][k0], 136, lane);
#pragma unroll
    for (int i = 0; i < 2; ++i)
#pragma unroll
      for (int j = 0; j < 2; ++j) acc[i][j] = wmma_bf16(af[i], bfr[j], acc[i][j]);
  }
  __syncthreads();

  // phase 2: Y_off operands (reuse LDS)
#pragma unroll 4
  for (int it = 0; it < 64; ++it) {              // SA[l][n] = C[l][n]*exp(Acs[l])
    int idx = it * 256 + tid;
    int l = idx >> 7, n = idx & 127;
    SA[l][n] = f2bf(bf2f(Cb[(tb + l) * 128 + n]) * expf(sAcs[l]));
  }
  long long sbo = ((long long)((dir * 32 + c) * 32 + h)) * 8192;
#pragma unroll 4
  for (int it = 0; it < 32; ++it) {              // SB[p][n] = Sin[n][p]
    int idx = it * 256 + tid;
    int n = idx >> 6, p = idx & 63;
    SB[p][n] = f2bf(Sin[sbo + (long long)n * 64 + p]);
  }
  __syncthreads();
#pragma unroll
  for (int k0 = 0; k0 < 128; k0 += 32) {
    v16bf af[2], bfr[2];
#pragma unroll
    for (int i = 0; i < 2; ++i) af[i] = frag_a(&SA[wm * 32 + i * 16][k0], 136, lane);
#pragma unroll
    for (int j = 0; j < 2; ++j) bfr[j] = frag_b(&SB[wn * 32 + j * 16][k0], 136, lane);
#pragma unroll
    for (int i = 0; i < 2; ++i)
#pragma unroll
      for (int j = 0; j < 2; ++j) acc[i][j] = wmma_bf16(af[i], bfr[j], acc[i][j]);
  }

  float dv = Dv[h];
#pragma unroll
  for (int i = 0; i < 2; ++i)
#pragma unroll
    for (int j = 0; j < 2; ++j) {
      int col = wn * 32 + j * 16 + (lane & 15);
#pragma unroll
      for (int r8 = 0; r8 < 8; ++r8) {
        int row = wm * 32 + i * 16 + r8 + 8 * (lane >> 4);
        int t = c * 128 + row;
        int tsrc = dir ? (SEQ - 1 - t) : t;
        float xv = xact[(long long)tsrc * CONVDIM + h * 64 + col];
        Yb[((long long)dir * SEQ + t) * 2048 + h * 64 + col] = acc[i][j][r8] + xv * dv;
      }
    }
}

// ---------------------------------------------------------------------------
// combine dirs (token-shift + un-flip) + fd skip + silu(z) gate + RMS norm
// one block per token
// ---------------------------------------------------------------------------
__global__ __launch_bounds__(256) void k_combine(const float* __restrict__ Yb,
                                                 const float* __restrict__ xact,
                                                 const float* __restrict__ zx,
                                                 const float* __restrict__ fdb,
                                                 const float* __restrict__ Dv,
                                                 const float* __restrict__ norm_w,
                                                 bh* __restrict__ yn_bf) {
  int t = blockIdx.x, tid = threadIdx.x;
  __shared__ float red[256];
  float yg[8];
  float ss = 0.f;
#pragma unroll
  for (int k = 0; k < 8; ++k) {
    int d = tid + k * 256;
    int h = d >> 6;
    float yf = (t >= 1) ? Yb[((long long)(t - 1)) * 2048 + d] : 0.f;
    float ybk = (t <= SEQ - 2) ? Yb[((long long)SEQ + (SEQ - 2 - t)) * 2048 + d] : 0.f;
    float xv = xact[(long long)t * CONVDIM + d];
    float fdv = fdb[(long long)t * 32 + h] + Dv[h];
    float v = yf + ybk + xv * fdv;
    float z = zx[(long long)t * DPROJ + d];
    float g = v * (z / (1.f + expf(-z)));
    yg[k] = g;
    ss += g * g;
  }
  red[tid] = ss;
  __syncthreads();
  for (int s = 128; s > 0; s >>= 1) {
    if (tid < s) red[tid] += red[tid + s];
    __syncthreads();
  }
  float rms = rsqrtf(red[0] / 2048.f + 1e-5f);
#pragma unroll
  for (int k = 0; k < 8; ++k) {
    int d = tid + k * 256;
    yn_bf[(long long)t * 2048 + d] = f2bf(yg[k] * rms * norm_w[d]);
  }
}

// ---------------------------------------------------------------------------
// host side
// ---------------------------------------------------------------------------
extern "C" void kernel_launch(void* const* d_in, const int* in_sizes, int n_in,
                              void* d_out, int out_size, void* d_ws, size_t ws_size,
                              hipStream_t stream) {
  (void)in_sizes; (void)n_in; (void)out_size; (void)ws_size;
  const float* u       = (const float*)d_in[0];
  const float* in_w    = (const float*)d_in[1];
  const float* conv_w  = (const float*)d_in[2];
  const float* conv_b  = (const float*)d_in[3];
  const float* dt_bias = (const float*)d_in[4];
  const float* A_log   = (const float*)d_in[5];
  const float* Dv      = (const float*)d_in[6];
  const float* fc_D_w  = (const float*)d_in[7];
  const float* norm_w  = (const float*)d_in[8];
  const float* out_w   = (const float*)d_in[9];
  float* out = (float*)d_out;

  char* cur = (char*)d_ws;
  auto alloc = [&](size_t bytes) -> void* {
    void* p = (void*)cur;
    cur += (bytes + 255) & ~(size_t)255;
    return p;
  };

  bh*    u_bf    = (bh*)   alloc((size_t)SEQ * DMODEL * 2);
  bh*    win_bf  = (bh*)   alloc((size_t)DPROJ * DMODEL * 2);
  bh*    wout_bf = (bh*)   alloc((size_t)DMODEL * DINNER * 2);
  bh*    fcw_bf  = (bh*)   alloc((size_t)NH * DINNER * 2);
  float* zx      = (float*)alloc((size_t)SEQ * DPROJ * 4);
  float* xact    = (float*)alloc((size_t)SEQ * CONVDIM * 4);
  float* dtb     = (float*)alloc((size_t)2 * SEQ * NH * 4);
  float* dtA     = (float*)alloc((size_t)2 * SEQ * NH * 4);
  float* Acs     = (float*)alloc((size_t)2 * NH * NCHUNK * CHUNKL * 4);
  float* csum    = (float*)alloc((size_t)2 * NH * NCHUNK * 4);
  bh*    xdt     = (bh*)   alloc((size_t)2 * SEQ * DINNER * 2);
  bh*    Bb      = (bh*)   alloc((size_t)2 * SEQ * DSTATE * 2);
  bh*    Cb      = (bh*)   alloc((size_t)2 * SEQ * DSTATE * 2);
  float* G       = (float*)alloc((size_t)2 * NCHUNK * CHUNKL * CHUNKL * 4);
  float* Sb      = (float*)alloc((size_t)2 * NCHUNK * NH * DSTATE * HD * 4);
  float* Sin     = (float*)alloc((size_t)2 * NCHUNK * NH * DSTATE * HD * 4);
  float* Yb      = (float*)alloc((size_t)2 * SEQ * DINNER * 4);
  bh*    xog_bf  = (bh*)   alloc((size_t)SEQ * DINNER * 2);
  float* fdb     = (float*)alloc((size_t)SEQ * NH * 4);
  bh*    yn_bf   = (bh*)   alloc((size_t)SEQ * DINNER * 2);

  dim3 blk(256);
  auto g1 = [](long long n) { return dim3((unsigned)((n + 255) / 256)); };

  // casts
  k_cast<<<g1((long long)SEQ * DMODEL), blk, 0, stream>>>(u, u_bf, SEQ, DMODEL, DMODEL, DMODEL);
  k_cast<<<g1((long long)DPROJ * DMODEL), blk, 0, stream>>>(in_w, win_bf, DPROJ, DMODEL, DMODEL, DMODEL);
  k_cast<<<g1((long long)DMODEL * DINNER), blk, 0, stream>>>(out_w, wout_bf, DMODEL, DINNER, DINNER, DINNER);
  k_cast<<<g1((long long)NH * DINNER), blk, 0, stream>>>(fc_D_w, fcw_bf, NH, DINNER, DINNER, DINNER);

  // in_proj:  zx = u @ in_w^T   (4096 x 4672 x 1024)
  k_gemm_nt<<<dim3((DPROJ + 127) / 128, SEQ / 128, 1), blk, 0, stream>>>(
      u_bf, win_bf, zx, SEQ, DPROJ, DMODEL, DMODEL, DMODEL, DPROJ, 0, 0, 0);

  // conv + silu
  k_conv<<<g1((long long)SEQ * CONVDIM), blk, 0, stream>>>(zx, conv_w, conv_b, xact);

  // dt / dtA, cumsums
  k_dt<<<g1(2LL * SEQ * NH), blk, 0, stream>>>(zx, dt_bias, A_log, dtb, dtA);
  k_cumsum<<<g1(2LL * NH * NCHUNK), blk, 0, stream>>>(dtA, Acs, csum);

  // per-direction arrays
  k_arrays<<<dim3(SEQ, 2), blk, 0, stream>>>(xact, dtb, xdt, Bb, Cb);

  // fd = x_og @ fc_D_w^T
  k_cast<<<g1((long long)SEQ * DINNER), blk, 0, stream>>>(xact, xog_bf, SEQ, DINNER, CONVDIM, DINNER);
  k_gemm_nt<<<dim3(1, SEQ / 128, 1), blk, 0, stream>>>(
      xog_bf, fcw_bf, fdb, SEQ, NH, DINNER, DINNER, DINNER, NH, 0, 0, 0);

  // G = C @ B^T per (dir,chunk): batched 128x128x128
  k_gemm_nt<<<dim3(1, 1, 2 * NCHUNK), blk, 0, stream>>>(
      Cb, Bb, G, CHUNKL, CHUNKL, DSTATE, DSTATE, DSTATE, CHUNKL,
      (long long)CHUNKL * DSTATE, (long long)CHUNKL * DSTATE,
      (long long)CHUNKL * CHUNKL);

  // SSD chunk pipeline
  k_states<<<dim3(NH, NCHUNK, 2), blk, 0, stream>>>(Bb, xdt, Acs, Sb);
  k_scan<<<dim3(NH, 2), blk, 0, stream>>>(Sb, csum, Sin);
  k_Y<<<dim3(NH, NCHUNK, 2), blk, 0, stream>>>(G, Cb, xdt, Acs, Sin, xact, Dv, Yb);

  // combine + gate + RMS norm
  k_combine<<<dim3(SEQ), blk, 0, stream>>>(Yb, xact, zx, fdb, Dv, norm_w, yn_bf);

  // out_proj: out = yn @ out_w^T  (4096 x 1024 x 2048)
  k_gemm_nt<<<dim3(DMODEL / 128, SEQ / 128, 1), blk, 0, stream>>>(
      yn_bf, wout_bf, out, SEQ, DMODEL, DINNER, DINNER, DINNER, DMODEL, 0, 0, 0);
}
